// DistributedGAT_82703890251956
// MI455X (gfx1250) — matrix operand order repaired and checked by reading, not compiled, not run on hardware
//
#include <hip/hip_runtime.h>
#include <hip/hip_bf16.h>
#include <hip/hip_fp16.h>

#define NNODES 20000
#define HEADS  4
#define FDIM   256   // H * C = 4 * 64, identical for all three layers
#define CHEAD  64

typedef __attribute__((ext_vector_type(16))) _Float16 v16h;
typedef __attribute__((ext_vector_type(8)))  _Float16 h8;
typedef __attribute__((ext_vector_type(8)))  float    v8f;

// ---- order-preserving float<->uint encoding for atomic segment-max --------
static __device__ __forceinline__ unsigned f2ord(float f) {
    unsigned u = __float_as_uint(f);
    return (u & 0x80000000u) ? ~u : (u | 0x80000000u);
}
static __device__ __forceinline__ float ord2f(unsigned e) {
    unsigned u = (e & 0x80000000u) ? (e & 0x7FFFFFFFu) : ~e;
    return __uint_as_float(u);
}
#define ENC_NEG_INF 0x007FFFFFu   // f2ord(-inf)

// ---------------------------------------------------------------- fills ----
__global__ void fill_u32(unsigned* __restrict__ p, unsigned v, int n) {
    int i = blockIdx.x * blockDim.x + threadIdx.x;
    if (i < n) p[i] = v;
}

// ------------------------------------------------------- f32 -> f16 conv ---
__global__ void conv_x(const float* __restrict__ x, _Float16* __restrict__ A, int n) {
    int i = blockIdx.x * blockDim.x + threadIdx.x;
    if (i < n) A[i] = (_Float16)x[i];
}

// W[K,256] f32 (row-major)  ->  Wt[256,K] f16 (transposed, row-major)
__global__ void conv_w(const float* __restrict__ W, _Float16* __restrict__ Wt, int K) {
    int i = blockIdx.x * blockDim.x + threadIdx.x;
    if (i >= K * FDIM) return;
    int k = i / FDIM, m = i % FDIM;
    Wt[m * K + k] = (_Float16)W[i];
}

// ----------------------------------------------------- WMMA GEMM ----------
// A: [NNODES,K] f16 row-major; Wt: [FDIM,K] f16 (B transposed); Hout: [NNODES,FDIM] f32
// One wave per 16x16 output tile; K-loop step 32 via v_wmma_f32_16x16x32_f16.
__global__ void __launch_bounds__(256)
gemm_wmma(const _Float16* __restrict__ A, const _Float16* __restrict__ Wt,
          float* __restrict__ Hout, int K) {
    const int lane = threadIdx.x & 31;
    const int wave = blockIdx.x * (blockDim.x >> 5) + (threadIdx.x >> 5);
    const int tilesM = FDIM / 16;                 // 16
    const int tm = wave % tilesM;
    const int tn = wave / tilesM;
    if (tn * 16 >= NNODES) return;

    const bool hi = lane >= 16;
    const int  r  = lane & 15;
    const _Float16* arow = A  + (size_t)(tn * 16 + r) * K + (hi ? 16 : 0);
    const _Float16* brow = Wt + (size_t)(tm * 16 + r) * K + (hi ? 16 : 0);

    v8f acc = {};
    for (int k0 = 0; k0 < K; k0 += 32) {
        h8 a_lo = *(const h8*)(arow + k0);        // K = k0+{0..7} (+16 if hi)
        h8 a_hi = *(const h8*)(arow + k0 + 8);    // K = k0+{8..15} (+16 if hi)
        v16h b  = *(const v16h*)(brow + k0);      // K = k0+{0..15} (+16 if hi)
        v16h a;
#pragma unroll
        for (int i = 0; i < 8; ++i) { a[i] = a_lo[i]; a[8 + i] = a_hi[i]; }
        acc = __builtin_amdgcn_wmma_f32_16x16x32_f16(
                  false, a, false, b, (short)0, acc, false, false);
    }
    float* crow = Hout + (size_t)(tn * 16 + (hi ? 8 : 0)) * FDIM + tm * 16 + r;
#pragma unroll
    for (int v = 0; v < 8; ++v) crow[(size_t)v * FDIM] = acc[v];
}

// ----------------------- per-node attention coefficients e_src/e_dst ------
// One wave per node. lane covers 8 contiguous channels; 8 lanes form one head.
__global__ void __launch_bounds__(256)
node_coeff(const float* __restrict__ H, const float* __restrict__ a_src,
           const float* __restrict__ a_dst, float* __restrict__ es,
           float* __restrict__ ed) {
    const int lane = threadIdx.x & 31;
    const int node = blockIdx.x * (blockDim.x >> 5) + (threadIdx.x >> 5);
    if (node >= NNODES) return;
    const int off  = lane * 8;          // 0..255
    const int head = off >> 6;
    const int c    = off & 63;
    const float4* hp = (const float4*)(H + (size_t)node * FDIM + off);
    float4 h0 = hp[0], h1 = hp[1];
    const float* sv = a_src + head * CHEAD + c;
    const float* dv = a_dst + head * CHEAD + c;
    float ss = h0.x*sv[0] + h0.y*sv[1] + h0.z*sv[2] + h0.w*sv[3]
             + h1.x*sv[4] + h1.y*sv[5] + h1.z*sv[6] + h1.w*sv[7];
    float ds = h0.x*dv[0] + h0.y*dv[1] + h0.z*dv[2] + h0.w*dv[3]
             + h1.x*dv[4] + h1.y*dv[5] + h1.z*dv[6] + h1.w*dv[7];
#pragma unroll
    for (int m = 1; m < 8; m <<= 1) {
        ss += __shfl_xor(ss, m, 32);
        ds += __shfl_xor(ds, m, 32);
    }
    if ((lane & 7) == 0) {
        es[node * HEADS + head] = ss;
        ed[node * HEADS + head] = ds;
    }
}

// ------------------- edge pass 1: leaky-relu logits + segment max ---------
__global__ void edge_logit_max(const int* __restrict__ ei, int E, int E2,
                               const float* __restrict__ es, const float* __restrict__ ed,
                               float* __restrict__ logit, unsigned* __restrict__ mx) {
    int e = blockIdx.x * blockDim.x + threadIdx.x;
    if (e >= E2) return;
    int s, d;
    if (e < E) { s = ei[e]; d = ei[E + e]; } else { s = d = e - E; }
    float4 a = *(const float4*)(es + (size_t)s * 4);
    float4 b = *(const float4*)(ed + (size_t)d * 4);
    float z[4] = { a.x + b.x, a.y + b.y, a.z + b.z, a.w + b.w };
    float4 lg;
    float* lp = (float*)&lg;
#pragma unroll
    for (int h = 0; h < 4; ++h) {
        float l = z[h] > 0.f ? z[h] : 0.2f * z[h];
        lp[h] = l;
        atomicMax(&mx[(size_t)d * 4 + h], f2ord(l));
    }
    *(float4*)(logit + (size_t)e * 4) = lg;
}

// ------------------- edge pass 2: exp(logit - max) + segment sum ----------
__global__ void edge_exp_sum(const int* __restrict__ ei, int E, int E2,
                             const unsigned* __restrict__ mx,
                             float* __restrict__ exl, float* __restrict__ den) {
    int e = blockIdx.x * blockDim.x + threadIdx.x;
    if (e >= E2) return;
    int d = (e < E) ? ei[E + e] : e - E;
    float4 lg = *(const float4*)(exl + (size_t)e * 4);
    float4 ex;
    ex.x = __expf(lg.x - ord2f(mx[(size_t)d * 4 + 0]));
    ex.y = __expf(lg.y - ord2f(mx[(size_t)d * 4 + 1]));
    ex.z = __expf(lg.z - ord2f(mx[(size_t)d * 4 + 2]));
    ex.w = __expf(lg.w - ord2f(mx[(size_t)d * 4 + 3]));
    *(float4*)(exl + (size_t)e * 4) = ex;
    atomicAdd(den + (size_t)d * 4 + 0, ex.x);
    atomicAdd(den + (size_t)d * 4 + 1, ex.y);
    atomicAdd(den + (size_t)d * 4 + 2, ex.z);
    atomicAdd(den + (size_t)d * 4 + 3, ex.w);
}

// ------------- edge pass 3: alpha output + weighted feature scatter -------
// One wave per edge; lane handles 8 contiguous channels (head = lane>>3).
// h and agg (20.5 MB each) live in the 192 MB L2 -> atomics stay on-chip.
__global__ void __launch_bounds__(256)
edge_agg(const int* __restrict__ ei, int E, int E2,
         const float* __restrict__ H, const float* __restrict__ exl,
         const float* __restrict__ den, float* __restrict__ agg,
         float* __restrict__ alpha_out) {
    const int lane = threadIdx.x & 31;
    const int e = blockIdx.x * (blockDim.x >> 5) + (threadIdx.x >> 5);
    if (e >= E2) return;
    int s, d;
    if (e < E) { s = ei[e]; d = ei[E + e]; } else { s = d = e - E; }
    const int head = lane >> 3;
    const float alpha = exl[(size_t)e * 4 + head] / den[(size_t)d * 4 + head];
    if (lane < 4)
        alpha_out[(size_t)e * 4 + lane] =
            exl[(size_t)e * 4 + lane] / den[(size_t)d * 4 + lane];
    const int off = lane * 8;
    const float4* hp = (const float4*)(H + (size_t)s * FDIM + off);
    float4 v0 = hp[0], v1 = hp[1];
    float* ap = agg + (size_t)d * FDIM + off;
    atomicAdd(ap + 0, alpha * v0.x);
    atomicAdd(ap + 1, alpha * v0.y);
    atomicAdd(ap + 2, alpha * v0.z);
    atomicAdd(ap + 3, alpha * v0.w);
    atomicAdd(ap + 4, alpha * v1.x);
    atomicAdd(ap + 5, alpha * v1.y);
    atomicAdd(ap + 6, alpha * v1.z);
    atomicAdd(ap + 7, alpha * v1.w);
}

// -------------------------- epilogues -------------------------------------
__global__ void finalize_elu(const float* __restrict__ agg, const float* __restrict__ bias,
                             _Float16* __restrict__ Anext, int n) {
    int i = blockIdx.x * blockDim.x + threadIdx.x;
    if (i >= n) return;
    float v = agg[i] + bias[i & (FDIM - 1)];
    v = v > 0.f ? v : expm1f(v);
    Anext[i] = (_Float16)v;
}

__global__ void finalize_mean(const float* __restrict__ agg, const float* __restrict__ b2,
                              float* __restrict__ out) {
    int i = blockIdx.x * blockDim.x + threadIdx.x;
    if (i >= NNODES * CHEAD) return;
    int n = i / CHEAD, c = i % CHEAD;
    const float* p = agg + (size_t)n * FDIM + c;
    out[i] = 0.25f * (p[0] + p[64] + p[128] + p[192]) + b2[c];
}

// ==========================================================================
extern "C" void kernel_launch(void* const* d_in, const int* in_sizes, int n_in,
                              void* d_out, int out_size, void* d_ws, size_t ws_size,
                              hipStream_t stream) {
    const float* x  = (const float*)d_in[0];
    const int*   ei = (const int*)d_in[1];
    const float* W[3]  = { (const float*)d_in[2], (const float*)d_in[6],  (const float*)d_in[10] };
    const float* As[3] = { (const float*)d_in[3], (const float*)d_in[7],  (const float*)d_in[11] };
    const float* Ad[3] = { (const float*)d_in[4], (const float*)d_in[8],  (const float*)d_in[12] };
    const float* Bs[3] = { (const float*)d_in[5], (const float*)d_in[9],  (const float*)d_in[13] };
    const int E  = in_sizes[1] / 2;
    const int E2 = E + NNODES;
    const int Kl[3] = { 128, 256, 256 };

    // workspace carve-up (256B aligned)
    char* wsb = (char*)d_ws;
    size_t off = 0;
    auto carve = [&](size_t bytes) -> void* {
        off = (off + 255) & ~(size_t)255;
        void* p = wsb + off;
        off += bytes;
        return p;
    };
    _Float16* Ah   = (_Float16*)carve((size_t)NNODES * FDIM * 2); // f16 activations
    _Float16* Wt   = (_Float16*)carve((size_t)FDIM * FDIM * 2);   // f16 W transposed
    float*    Hbuf = (float*)   carve((size_t)NNODES * FDIM * 4); // GEMM output
    float*    es   = (float*)   carve((size_t)NNODES * 4 * 4);
    float*    ed   = (float*)   carve((size_t)NNODES * 4 * 4);
    unsigned* mx   = (unsigned*)carve((size_t)NNODES * 4 * 4);
    float*    den  = (float*)   carve((size_t)NNODES * 4 * 4);
    float*    exl  = (float*)   carve((size_t)E2 * 4 * 4);
    float*    agg  = (float*)   carve((size_t)NNODES * FDIM * 4);

    float* out = (float*)d_out;
    float* alpha_base = out + (size_t)NNODES * CHEAD;

    conv_x<<<(NNODES * 128 + 255) / 256, 256, 0, stream>>>(x, Ah, NNODES * 128);

    for (int L = 0; L < 3; ++L) {
        const int K = Kl[L];
        conv_w<<<(K * FDIM + 255) / 256, 256, 0, stream>>>(W[L], Wt, K);
        fill_u32<<<(NNODES * FDIM + 255) / 256, 256, 0, stream>>>((unsigned*)agg, 0u, NNODES * FDIM);
        fill_u32<<<(NNODES * 4 + 255) / 256, 256, 0, stream>>>((unsigned*)den, 0u, NNODES * 4);
        fill_u32<<<(NNODES * 4 + 255) / 256, 256, 0, stream>>>(mx, ENC_NEG_INF, NNODES * 4);

        const int waves = (NNODES / 16) * (FDIM / 16);   // 20000 tiles
        gemm_wmma<<<waves / 8, 256, 0, stream>>>(Ah, Wt, Hbuf, K);

        node_coeff<<<(NNODES + 7) / 8, 256, 0, stream>>>(Hbuf, As[L], Ad[L], es, ed);
        edge_logit_max<<<(E2 + 255) / 256, 256, 0, stream>>>(ei, E, E2, es, ed, exl, mx);
        edge_exp_sum<<<(E2 + 255) / 256, 256, 0, stream>>>(ei, E, E2, mx, exl, den);
        edge_agg<<<(E2 + 7) / 8, 256, 0, stream>>>(ei, E, E2, Hbuf, exl, den, agg,
                                                   alpha_base + (size_t)L * E2 * 4);
        if (L < 2)
            finalize_elu<<<(NNODES * FDIM + 255) / 256, 256, 0, stream>>>(agg, Bs[L], Ah, NNODES * FDIM);
        else
            finalize_mean<<<(NNODES * CHEAD + 255) / 256, 256, 0, stream>>>(agg, Bs[2], out);
    }
}
